// VectorQuantizerEMA_15281493639807
// MI455X (gfx1250) — matrix operand order repaired and checked by reading, not compiled, not run on hardware
//
#include <hip/hip_runtime.h>
#include <hip/hip_bf16.h>

// ---------------- Problem constants ----------------
#define BATCH 32
#define CHAN  80
#define LEN   4096
#define NCODE 512
#define NROWS (BATCH * LEN)          // 131072 flattened vectors
#define NELEM (BATCH * CHAN * LEN)   // 10485760 elements of x / quantized
#define KSTEPS 3                     // ceil(80 / 32) k-steps of 32 channels

typedef __attribute__((ext_vector_type(16))) __bf16          v16bf;
typedef __attribute__((ext_vector_type(16))) unsigned short  v16u;
typedef __attribute__((ext_vector_type(8)))  float           v8f;

// ---------------- Workspace layout (bytes) ----------------
// B fragments: 32 code-tiles x 3 ksteps x 2 (hi/lo) x 32 lanes x 16 bf16
#define BFRAG_BYTES (32u * KSTEPS * 2u * 32u * 32u)  // 196608
#define OFF_BFRAG 0u
#define OFF_WSQ   (OFF_BFRAG + BFRAG_BYTES)          // 196608, 512 floats
#define OFF_IDX   (OFF_WSQ + 2048u)                  // 198656, 131072 ints
#define OFF_HIST  (OFF_IDX + 4u * NROWS)             // 722944, 512 uints
#define OFF_SSE   (OFF_HIST + 2048u)                 // 724992, 1 float

// RNE float -> bf16 bits (finite inputs)
static __device__ __forceinline__ unsigned short f2bf(float f) {
  union { float f; unsigned u; } v; v.f = f;
  unsigned u = v.u;
  return (unsigned short)((u + 0x7FFFu + ((u >> 16) & 1u)) >> 16);
}
static __device__ __forceinline__ float bf2f(unsigned short h) {
  union { unsigned u; float f; } v; v.u = ((unsigned)h) << 16;
  return v.f;
}

// ---------------------------------------------------------------------------
// Kernel 0: pre-swizzle codebook into WMMA-B fragment order (bf16 hi/lo),
// compute ||w_k||^2, zero histogram + SSE accumulator.
// B fragment layout this code uses (contiguous k, same as A): lane n = lane%16 selects the code
// column, K = (lane/16)*16 + e selects the channel within the 32-wide k-step.
// ---------------------------------------------------------------------------
__global__ void __launch_bounds__(256) vq_prep(const float* __restrict__ weight,
                                               unsigned char* __restrict__ ws) {
  int tid = blockIdx.x * blockDim.x + threadIdx.x;
  unsigned short* bfrag = (unsigned short*)(ws + OFF_BFRAG);
  float*    wsq  = (float*)(ws + OFF_WSQ);
  unsigned* hist = (unsigned*)(ws + OFF_HIST);
  float*    sse  = (float*)(ws + OFF_SSE);

  if (tid < 32 * KSTEPS * 2 * 32) {           // 6144 fragment-lane slots
    int lane = tid & 31;
    int p    = (tid >> 5) & 1;                // 0 = hi, 1 = lo
    int s    = (tid >> 6) % KSTEPS;           // k-step
    int j    = tid / (64 * KSTEPS);           // code tile (16 codes)
    int code  = (j << 4) + (lane & 15);
    int cbase = s * 32 + ((lane >> 4) << 4);
    #pragma unroll
    for (int e = 0; e < 16; ++e) {
      int c = cbase + e;
      float wv = (c < CHAN) ? weight[code * CHAN + c] : 0.0f;
      unsigned short hi = f2bf(wv);
      bfrag[tid * 16 + e] = (p == 0) ? hi : f2bf(wv - bf2f(hi));
    }
  }
  if (tid < NCODE) {
    float s = 0.0f;
    for (int c = 0; c < CHAN; ++c) {
      float w = weight[tid * CHAN + c];
      s = fmaf(w, w, s);
    }
    wsq[tid]  = s;
    hist[tid] = 0u;
  }
  if (tid == 0) *sse = 0.0f;
}

// ---------------------------------------------------------------------------
// Kernel 1: nearest-code search.  Each wave owns 16 consecutive rows (one
// 16x96 A tile, bf16 hi/lo split, kept in VGPRs) and sweeps all 512 codes.
// dist[n,k] = ||w_k||^2 - 2 * (x_n . w_k); dot via 9 bf16 WMMAs per tile
// (hi*hi + hi*lo + lo*hi) so the result carries ~fp32 accuracy.
// ---------------------------------------------------------------------------
__global__ void __launch_bounds__(256) vq_argmin(const float* __restrict__ x,
                                                 unsigned char* __restrict__ ws) {
  __shared__ uint4 ldsB[3072];     // 48 KB: 8 code tiles of B fragments
  __shared__ float wsqs[NCODE];    // 2 KB

  const unsigned short* bfragG = (const unsigned short*)(ws + OFF_BFRAG);
  const float* wsqG = (const float*)(ws + OFF_WSQ);
  int*      idx  = (int*)(ws + OFF_IDX);
  unsigned* hist = (unsigned*)(ws + OFF_HIST);

  const int t    = threadIdx.x;
  const int wv_  = t >> 5;
  const int lane = t & 31;
  wsqs[t]       = wsqG[t];
  wsqs[t + 256] = wsqG[t + 256];

  const int n0 = blockIdx.x * 128 + wv_ * 16;   // 16 rows per wave
  const int b  = n0 >> 12;                      // batch (tiles never cross b)
  const int l0 = n0 & 4095;
  const int m  = lane & 15;                     // row within tile / column idx
  const int h  = lane >> 4;
  const float* xb = x + (size_t)b * (CHAN * LEN) + (l0 + m);

  // A fragments: 16-bit A 16x32 layout -> lane m, element e holds K(e,h)
  v16bf Ah[KSTEPS], Al[KSTEPS];
  for (int s = 0; s < KSTEPS; ++s) {
    v16u th, tl;
    #pragma unroll
    for (int e = 0; e < 16; ++e) {
      int v  = e >> 1;
      int kk = (h << 4) + e + (v & 0);
      int c  = s * 32 + kk;
      float xv = (c < CHAN) ? xb[(size_t)c * LEN] : 0.0f;
      unsigned short hi = f2bf(xv);
      th[e] = hi;
      tl[e] = f2bf(xv - bf2f(hi));
    }
    Ah[s] = __builtin_bit_cast(v16bf, th);
    Al[s] = __builtin_bit_cast(v16bf, tl);
  }

  float best[8];
  int   bestk[8];
  #pragma unroll
  for (int r = 0; r < 8; ++r) { best[r] = 3.4e38f; bestk[r] = 0; }

  for (int cc = 0; cc < 4; ++cc) {            // 4 chunks of 128 codes
    __syncthreads();
    const uint4* src = (const uint4*)bfragG + cc * 3072;
    for (int i = t; i < 3072; i += 256) ldsB[i] = src[i];
    __syncthreads();

    const v16bf* bp = (const v16bf*)ldsB;
    for (int jl = 0; jl < 8; ++jl) {          // 8 code tiles per chunk
      const int j = cc * 8 + jl;
      v8f acc = {0.f, 0.f, 0.f, 0.f, 0.f, 0.f, 0.f, 0.f};
      #pragma unroll
      for (int s = 0; s < KSTEPS; ++s) {
        v16bf bh = bp[(((jl * KSTEPS + s) << 1) + 0) * 32 + lane];
        v16bf bl = bp[(((jl * KSTEPS + s) << 1) + 1) * 32 + lane];
        acc = __builtin_amdgcn_wmma_f32_16x16x32_bf16(false, Ah[s], false, bh,
                                                      (short)0, acc, false, false);
        acc = __builtin_amdgcn_wmma_f32_16x16x32_bf16(false, Ah[s], false, bl,
                                                      (short)0, acc, false, false);
        acc = __builtin_amdgcn_wmma_f32_16x16x32_bf16(false, Al[s], false, bh,
                                                      (short)0, acc, false, false);
      }
      // C layout: lane column N = m; VGPR r holds row M = r + 8*h
      const int   code = (j << 4) + m;
      const float wq   = wsqs[code];
      #pragma unroll
      for (int r = 0; r < 8; ++r) {
        float d = fmaf(-2.0f, acc[r], wq);
        if (d < best[r]) { best[r] = d; bestk[r] = code; }
      }
    }
  }

  // Cross-lane argmin over the 16 columns held per row group (ties -> low k)
  #pragma unroll
  for (int off = 1; off < 16; off <<= 1) {
    #pragma unroll
    for (int r = 0; r < 8; ++r) {
      float od = __shfl_xor(best[r], off, 32);
      int   ok = __shfl_xor(bestk[r], off, 32);
      if (od < best[r] || (od == best[r] && ok < bestk[r])) {
        best[r] = od; bestk[r] = ok;
      }
    }
  }
  if (m == 0) {                               // lanes 0 and 16
    #pragma unroll
    for (int r = 0; r < 8; ++r) {
      int row = r + (h << 3);
      idx[n0 + row] = bestk[r];
      atomicAdd(&hist[bestk[r]], 1u);
    }
  }
}

// ---------------------------------------------------------------------------
// Kernel 2: gather winning codes, emit quantized_st = x + (q - x) in [B,C,L]
// layout, accumulate SSE for the loss (block-reduced, one atomic per block).
// ---------------------------------------------------------------------------
__global__ void __launch_bounds__(256) vq_quantize(const float* __restrict__ x,
                                                   const float* __restrict__ weight,
                                                   float* __restrict__ out,
                                                   unsigned char* __restrict__ ws) {
  const int* idx = (const int*)(ws + OFF_IDX);
  float* sse = (float*)(ws + OFF_SSE);

  int tid = blockIdx.x * 256 + threadIdx.x;   // linear over [B,C,L]
  int l  = tid & 4095;
  int bc = tid >> 12;                         // b*CHAN + c
  int c  = bc % CHAN;
  int b  = bc / CHAN;
  int n  = (b << 12) + l;

  int   k    = idx[n];
  float q    = weight[k * CHAN + c];
  float xv   = x[tid];
  float diff = q - xv;
  out[1 + tid] = xv + diff;                   // straight-through value

  __shared__ float red[256];
  red[threadIdx.x] = diff * diff;
  __syncthreads();
  #pragma unroll
  for (int s = 128; s > 0; s >>= 1) {
    if (threadIdx.x < s) red[threadIdx.x] += red[threadIdx.x + s];
    __syncthreads();
  }
  if (threadIdx.x == 0) atomicAdd(sse, red[0]);
}

// ---------------------------------------------------------------------------
// Kernel 3: loss scalar + perplexity from the histogram.
// ---------------------------------------------------------------------------
__global__ void __launch_bounds__(512) vq_finalize(float* __restrict__ out,
                                                   int out_size,
                                                   unsigned char* __restrict__ ws) {
  const unsigned* hist = (const unsigned*)(ws + OFF_HIST);
  const float*    sse  = (const float*)(ws + OFF_SSE);
  __shared__ float red[512];
  int t = threadIdx.x;
  float p = (float)hist[t] / (float)NROWS;
  red[t] = p * logf(p + 1e-10f);
  __syncthreads();
  #pragma unroll
  for (int s = 256; s > 0; s >>= 1) {
    if (t < s) red[t] += red[t + s];
    __syncthreads();
  }
  if (t == 0) {
    out[0] = 1.25f * (*sse) / (float)NELEM;   // q_latent + 0.25 * e_latent
    out[out_size - 1] = expf(-red[0]);        // perplexity
  }
}

// ---------------------------------------------------------------------------
extern "C" void kernel_launch(void* const* d_in, const int* in_sizes, int n_in,
                              void* d_out, int out_size, void* d_ws, size_t ws_size,
                              hipStream_t stream) {
  const float* x      = (const float*)d_in[0];   // [32, 80, 4096]
  const float* weight = (const float*)d_in[1];   // [512, 80]
  float* out = (float*)d_out;                    // [1 + 10485760 + 1]
  unsigned char* ws = (unsigned char*)d_ws;

  vq_prep    <<<24,          256, 0, stream>>>(weight, ws);
  vq_argmin  <<<NROWS / 128, 256, 0, stream>>>(x, ws);
  vq_quantize<<<NELEM / 256, 256, 0, stream>>>(x, weight, out, ws);
  vq_finalize<<<1,           512, 0, stream>>>(out, out_size, ws);
}
